// CausalSelfAttention_5257039970399
// MI455X (gfx1250) — matrix operand order repaired and checked by reading, not compiled, not run on hardware
//
#include <hip/hip_runtime.h>

// ---------------------------------------------------------------------------
// CausalSelfAttention (c_attn -> causal average -> c_proj) for MI455X gfx1250
// B=4, T=2048, C=1024, H=16. Memory-bound: bf16 WMMA GEMMs + prefix-scan.
// ---------------------------------------------------------------------------

typedef __attribute__((ext_vector_type(16))) __bf16 v16bf;
typedef __attribute__((ext_vector_type(8)))  __bf16 v8bf;
typedef __attribute__((ext_vector_type(8)))  float  v8f;

#define BT   8192      // B*T rows
#define CDIM 1024      // channels

// round-to-nearest-even fp32 -> bf16 (bit pattern)
__device__ __forceinline__ unsigned short f2bf(float f) {
    unsigned int u = __builtin_bit_cast(unsigned int, f);
    u += 0x7FFFu + ((u >> 16) & 1u);
    return (unsigned short)(u >> 16);
}

// ---------------------------------------------------------------------------
// 1) fp32 -> bf16 elementwise convert (float4 vectorized)
// ---------------------------------------------------------------------------
__global__ void cvt_f32_bf16(const float* __restrict__ src,
                             unsigned short* __restrict__ dst, int n4) {
    int i = blockIdx.x * blockDim.x + threadIdx.x;
    if (i >= n4) return;
    float4 f = ((const float4*)src)[i];
    ushort4 o;
    o.x = f2bf(f.x); o.y = f2bf(f.y); o.z = f2bf(f.z); o.w = f2bf(f.w);
    ((ushort4*)dst)[i] = o;
}

// ---------------------------------------------------------------------------
// 2) transpose + convert:  Wt[c][d] = bf16(W[d][c]),  N x N, LDS 32x32 tiles
// ---------------------------------------------------------------------------
__global__ void transpose_cvt(const float* __restrict__ W,
                              unsigned short* __restrict__ Wt, int N) {
    __shared__ unsigned short tile[32][33];   // +1 pad: avoid bank conflicts
    int tx = threadIdx.x & 31;
    int ty = threadIdx.x >> 5;                 // 0..7, 256 threads
    int bx = blockIdx.x * 32;                  // col (c) block in W
    int by = blockIdx.y * 32;                  // row (d) block in W
#pragma unroll
    for (int j = 0; j < 4; ++j) {
        int r = ty + j * 8;
        tile[r][tx] = f2bf(W[(size_t)(by + r) * N + bx + tx]);
    }
    __syncthreads();
#pragma unroll
    for (int j = 0; j < 4; ++j) {
        int r = ty + j * 8;
        // Wt[bx+r][by+tx] = W[by+tx][bx+r]
        Wt[(size_t)(bx + r) * N + by + tx] = tile[tx][r];
    }
}

// ---------------------------------------------------------------------------
// WMMA fragment loaders (CDNA5 v_wmma_f32_16x16x32_bf16 layouts, ISA 7.12.2)
// A 16x32 bf16: lane m=lane&15; elems 0-7 -> K base (lo?0:8), 8-15 -> (lo?16:24)
// B 32x16 bf16: lane n=lane&15; elems 0-7 -> K base (lo?0:8), 8-15 -> (lo?16:24)
// ---------------------------------------------------------------------------
__device__ __forceinline__ v16bf load_a_frag(const unsigned short* __restrict__ A,
                                             int K, int mtile, int kk, int lane) {
    const bool lo = lane < 16;
    const int  m  = mtile + (lane & 15);
    const __bf16* base = (const __bf16*)(A + (size_t)m * K + kk);
    v8bf x0 = *(const v8bf*)(base + (lo ? 0 : 8));
    v8bf x1 = *(const v8bf*)(base + (lo ? 16 : 24));
    return __builtin_shufflevector(x0, x1, 0,1,2,3,4,5,6,7,8,9,10,11,12,13,14,15);
}

__device__ __forceinline__ v16bf load_b_frag(const unsigned short* __restrict__ Bt,
                                             int N, int kk, int ntile, int lane) {
    const int kb = kk + (lane < 16 ? 0 : 8);
    const __bf16* base = (const __bf16*)(Bt + (size_t)kb * N + ntile + (lane & 15));
    v16bf r;
#pragma unroll
    for (int s = 0; s < 8; ++s) {
        r[s]     = base[(size_t)s * N];
        r[s + 8] = base[(size_t)(s + 16) * N];
    }
    return r;
}

// ---------------------------------------------------------------------------
// 3/5) GEMM  C[M,N] (f32) = A[M,K] (bf16) * Bt[K,N] (bf16, pre-transposed W)
// One wave per 32x32 output tile: 2x2 accumulators, 4 WMMAs per K-step.
// blockDim = 128 (4 waves -> 4 consecutive N tiles), grid = (M/32, N/128).
// ---------------------------------------------------------------------------
__global__ void __launch_bounds__(128)
gemm_bf16_wmma(const unsigned short* __restrict__ A,
               const unsigned short* __restrict__ Bt,
               float* __restrict__ C, int M, int N, int K) {
    const int lane  = threadIdx.x & 31;
    const int wave  = threadIdx.x >> 5;
    const int mbase = blockIdx.x * 32;
    const int nbase = blockIdx.y * 128 + wave * 32;

    v8f acc00 = {}, acc01 = {}, acc10 = {}, acc11 = {};

    for (int kk = 0; kk < K; kk += 32) {
        v16bf a0 = load_a_frag(A, K, mbase,      kk, lane);
        v16bf a1 = load_a_frag(A, K, mbase + 16, kk, lane);
        v16bf b0 = load_b_frag(Bt, N, kk, nbase,      lane);
        v16bf b1 = load_b_frag(Bt, N, kk, nbase + 16, lane);
        acc00 = __builtin_amdgcn_wmma_f32_16x16x32_bf16(false, a0, false, b0,
                                                        (short)0, acc00, false, false);
        acc01 = __builtin_amdgcn_wmma_f32_16x16x32_bf16(false, a0, false, b1,
                                                        (short)0, acc01, false, false);
        acc10 = __builtin_amdgcn_wmma_f32_16x16x32_bf16(false, a1, false, b0,
                                                        (short)0, acc10, false, false);
        acc11 = __builtin_amdgcn_wmma_f32_16x16x32_bf16(false, a1, false, b1,
                                                        (short)0, acc11, false, false);
    }

    // C/D layout: VGPR r -> row (r + (lane<16?0:8)), col = lane&15
    const int mrow = (lane < 16 ? 0 : 8);
    const int ncol = lane & 15;
#pragma unroll
    for (int r = 0; r < 8; ++r) {
        size_t r0 = (size_t)(mbase + r + mrow) * N;
        size_t r1 = (size_t)(mbase + 16 + r + mrow) * N;
        C[r0 + nbase + ncol]      = acc00[r];
        C[r0 + nbase + 16 + ncol] = acc01[r];
        C[r1 + nbase + ncol]      = acc10[r];
        C[r1 + nbase + 16 + ncol] = acc11[r];
    }
}

// ---------------------------------------------------------------------------
// 4) causal running average along T per (b, c):
//    y[b,t,c] = (1/(t+1)) * sum_{s<=t} v[b,s,c];  writes bf16 for GEMM2.
//    4096 threads total, coalesced across c at each t.
// ---------------------------------------------------------------------------
__global__ void causal_avg_scan(const float* __restrict__ V,
                                unsigned short* __restrict__ Y) {
    int tid = blockIdx.x * blockDim.x + threadIdx.x;   // 0..4095
    int b = tid >> 10;
    int c = tid & (CDIM - 1);
    const float* p = V + (size_t)b * 2048 * CDIM + c;
    unsigned short* q = Y + (size_t)b * 2048 * CDIM + c;
    float sum = 0.f;
    for (int t = 0; t < 2048; ++t) {
        sum += p[(size_t)t * CDIM];
        q[(size_t)t * CDIM] = f2bf(sum / (float)(t + 1));
    }
}

// ---------------------------------------------------------------------------
// launch
// ---------------------------------------------------------------------------
extern "C" void kernel_launch(void* const* d_in, const int* in_sizes, int n_in,
                              void* d_out, int out_size, void* d_ws, size_t ws_size,
                              hipStream_t stream) {
    const float* x  = (const float*)d_in[0];   // [4,2048,1024]
    const float* Wc = (const float*)d_in[1];   // [1024,1024]
    const float* Wp = (const float*)d_in[2];   // [1024,1024]
    float* out = (float*)d_out;                // [4,2048,1024]

    // workspace layout
    unsigned short* Xbf = (unsigned short*)d_ws;                 // 16 MB
    unsigned short* Ybf = Xbf + (size_t)BT * CDIM;               // 16 MB
    unsigned short* WcT = Ybf + (size_t)BT * CDIM;               //  2 MB
    unsigned short* WpT = WcT + (size_t)CDIM * CDIM;             //  2 MB
    float*          V   = (float*)(WpT + (size_t)CDIM * CDIM);   // 32 MB

    // 1) x -> bf16
    {
        int n4 = (BT * CDIM) / 4;            // 2,097,152
        cvt_f32_bf16<<<(n4 + 255) / 256, 256, 0, stream>>>(x, Xbf, n4);
    }
    // 2) weight transposes (bf16)
    {
        dim3 g(CDIM / 32, CDIM / 32);
        transpose_cvt<<<g, 256, 0, stream>>>(Wc, WcT, CDIM);
        transpose_cvt<<<g, 256, 0, stream>>>(Wp, WpT, CDIM);
    }
    // 3) V = X * Wc^T
    {
        dim3 g(BT / 32, CDIM / 128);
        gemm_bf16_wmma<<<g, 128, 0, stream>>>(Xbf, WcT, V, BT, CDIM, CDIM);
    }
    // 4) causal averaging scan -> bf16
    causal_avg_scan<<<16, 256, 0, stream>>>(V, Ybf);
    // 5) out = Y * Wp^T
    {
        dim3 g(BT / 32, CDIM / 128);
        gemm_bf16_wmma<<<g, 128, 0, stream>>>(Ybf, WpT, out, BT, CDIM, CDIM);
    }
}
